// Attention_CNNE_36575941492780
// MI455X (gfx1250) — matrix operand order repaired and checked by reading, not compiled, not run on hardware
//
#include <hip/hip_runtime.h>
#include <hip/hip_bf16.h>
#include <math.h>

// ---------------- problem constants (match reference) ----------------
#define BATCH   64
#define CIN     32          // 2*FEATURE_NUM
#define T_IN    2048
#define FILTERS 64
#define WINDOW  5
#define TCONV1  2044        // T_IN - WINDOW + 1
#define TPOOL   2040        // TCONV1 - WINDOW + 1
#define L_OUT   2036        // TPOOL - WINDOW + 1
#define D_EMB   256
#define DA      64
#define SECTORS 11
#define KPAD    2048        // L_OUT padded to multiple of 32 for WMMA K
#define NCHUNK  (KPAD / 32)
#define BN_EPS  1e-5f

// ---------------- workspace layout (bytes from d_ws) ----------------
// h1:    [BATCH][FILTERS][TPOOL] f32
// stats: [0..63]=sum, [64..127]=sumsq, [128..191]=scale, [192]=cshift  (256 f32 reserved)
// idx:   [BATCH][L_OUT] i32
// W1h:   [DA][KPAD] f16 (zero padded K>=L_OUT)
#define H1_FLOATS    ((size_t)BATCH * FILTERS * TPOOL)           // 8,355,840 f32
#define STATS_FLOFF  H1_FLOATS
#define IDX_BYTE_OFF (((size_t)STATS_FLOFF + 256) * 4)
#define W1H_BYTE_OFF (IDX_BYTE_OFF + (size_t)BATCH * L_OUT * 4)

typedef _Float16 h8   __attribute__((ext_vector_type(8)));
typedef _Float16 v16h __attribute__((ext_vector_type(16)));
typedef float    v8f  __attribute__((ext_vector_type(8)));

// ================= kernel 1: W1 -> f16 (K padded), zero stats =================
__global__ void k_init(const float* __restrict__ W1, _Float16* __restrict__ W1h,
                       float* __restrict__ stats) {
    int i = blockIdx.x * blockDim.x + threadIdx.x;
    if (i < DA * KPAD) {
        int m = i >> 11;            // KPAD == 2048
        int k = i & (KPAD - 1);
        W1h[i] = (k < L_OUT) ? (_Float16)W1[m * L_OUT + k] : (_Float16)0.0f;
    }
    if (i < 128) stats[i] = 0.0f;   // sum + sumsq accumulators
}

// ========== kernel 2: conv1 + bias -> maxpool(5) -> per-channel stats ==========
__global__ __launch_bounds__(256)
void k_conv1_pool_stats(const float* __restrict__ x, const float* __restrict__ w1,
                        const float* __restrict__ b1, float* __restrict__ h1,
                        float* __restrict__ stats) {
    __shared__ float xs[CIN * 264];          // input strip, 264 = 256 + 2*(WINDOW-1)
    __shared__ float wl[CIN * WINDOW];
    __shared__ float cbuf[260];              // conv outputs for this tile
    __shared__ float red[256];
    const int tid  = threadIdx.x;
    const int tile = blockIdx.x;             // 8 tiles of 256 pooled outputs
    const int f    = blockIdx.y;
    const int b    = blockIdx.z;
    const int p0   = tile * 256;

    if (tid < CIN * WINDOW) wl[tid] = w1[f * CIN * WINDOW + tid];
    for (int i = tid; i < CIN * 264; i += 256) {
        int c = i / 264, o = i - c * 264;
        int t = p0 + o;
        xs[i] = (t < T_IN) ? x[((size_t)b * CIN + c) * T_IN + t] : 0.0f;
    }
    __syncthreads();

    const float bias = b1[f];
    for (int ti = tid; ti < 260; ti += 256) {
        float acc = bias;
        #pragma unroll
        for (int c = 0; c < CIN; ++c) {
            const float* xr = &xs[c * 264 + ti];
            const float* wr = &wl[c * WINDOW];
            #pragma unroll
            for (int k = 0; k < WINDOW; ++k) acc += xr[k] * wr[k];
        }
        cbuf[ti] = acc;                      // values with t>=TCONV1 never used by valid p
    }
    __syncthreads();

    const int p = p0 + tid;
    float m = 0.f, s1 = 0.f, s2 = 0.f;
    if (p < TPOOL) {
        m = cbuf[tid];
        #pragma unroll
        for (int k = 1; k < WINDOW; ++k) m = fmaxf(m, cbuf[tid + k]);
        h1[((size_t)b * FILTERS + f) * TPOOL + p] = m;
        s1 = m; s2 = m * m;
    }
    red[tid] = s1; __syncthreads();
    for (int s = 128; s > 0; s >>= 1) { if (tid < s) red[tid] += red[tid + s]; __syncthreads(); }
    if (tid == 0) atomicAdd(&stats[f], red[0]);
    __syncthreads();
    red[tid] = s2; __syncthreads();
    for (int s = 128; s > 0; s >>= 1) { if (tid < s) red[tid] += red[tid + s]; __syncthreads(); }
    if (tid == 0) atomicAdd(&stats[64 + f], red[0]);
}

// ====== kernel 3: BN scale per channel + fold shift/conv2-bias into cshift ======
__global__ void k_bn_finalize(float* __restrict__ stats, const float* __restrict__ gamma,
                              const float* __restrict__ beta, const float* __restrict__ w2,
                              const float* __restrict__ b2) {
    __shared__ float red[64];
    const int f = threadIdx.x;               // 64 threads
    const float invN = 1.0f / (float)(BATCH * TPOOL);
    float mu  = stats[f] * invN;
    float var = stats[64 + f] * invN - mu * mu;   // biased, as torch train-mode BN
    float rs  = rsqrtf(var + BN_EPS);
    float sc  = gamma[f] * rs;
    float sh  = beta[f] - mu * sc;
    stats[128 + f] = sc;
    float w2sum = 0.f;
    #pragma unroll
    for (int k = 0; k < WINDOW; ++k) w2sum += w2[f * WINDOW + k];
    red[f] = sh * w2sum;                     // time-invariant shift contribution
    __syncthreads();
    for (int s = 32; s > 0; s >>= 1) { if (f < s) red[f] += red[f + s]; __syncthreads(); }
    if (f == 0) stats[192] = red[0] + b2[0];
}

// ====== kernel 4: (h1*scale) conv2 + cshift -> relu -> sigmoid -> bucket idx ======
__global__ __launch_bounds__(256)
void k_conv2_sigmoid_idx(const float* __restrict__ h1, const float* __restrict__ w2,
                         const float* __restrict__ stats, int* __restrict__ idxb) {
    __shared__ float hs[32 * 264];           // half of the channels at a time
    __shared__ float w2s[FILTERS * WINDOW];
    __shared__ float sc[FILTERS];
    const int tid  = threadIdx.x;
    const int tile = blockIdx.x;
    const int b    = blockIdx.y;
    const int l0   = tile * 256;

    if (tid < FILTERS * WINDOW) w2s[tid] = w2[tid];
    if (tid < FILTERS)          sc[tid]  = stats[128 + tid];
    __syncthreads();

    float acc = stats[192];                  // folded BN shift + conv2 bias
    for (int half = 0; half < 2; ++half) {
        for (int i = tid; i < 32 * 264; i += 256) {
            int fi = i / 264, o = i - fi * 264;
            int f  = half * 32 + fi;
            int l  = l0 + o;
            hs[i] = (l < TPOOL) ? h1[((size_t)b * FILTERS + f) * TPOOL + l] * sc[f] : 0.f;
        }
        __syncthreads();
        if (l0 + tid < L_OUT) {
            #pragma unroll
            for (int fi = 0; fi < 32; ++fi) {
                const float* hr = &hs[fi * 264 + tid];
                const float* wr = &w2s[(half * 32 + fi) * WINDOW];
                #pragma unroll
                for (int k = 0; k < WINDOW; ++k) acc += hr[k] * wr[k];
            }
        }
        __syncthreads();
    }
    const int l = l0 + tid;
    if (l < L_OUT) {
        float r = fmaxf(acc, 0.f);                     // relu
        float c = 1.0f / (1.0f + expf(-r));            // sigmoid
        int id = (int)ceilf(c * 256.0f) - 1;           // first j with c <= (j+1)/D
        id = id < 0 ? 0 : (id > (D_EMB - 1) ? (D_EMB - 1) : id);
        idxb[b * L_OUT + l] = id;
    }
}

// ====== kernel 5: fused gather + WMMA GEMM + tanh + attention + softmax ======
// One block (8 wave32) per batch sample.
//  GEMM: t[64,256] = W1h[64,KPAD] x E[KPAD,256] with E[k,d] = f16(emb[idx[k]][d])
//  wave w: M-tile = w&3, N-tiles = (w>>2)*8 .. +7  (4x16 = 64 16x16 tiles)
__global__ __launch_bounds__(256)
void k_attention(const float* __restrict__ emb, const float* __restrict__ W2,
                 const float* __restrict__ lin_w, const float* __restrict__ lin_b,
                 const int* __restrict__ idxb, const _Float16* __restrict__ W1h,
                 float* __restrict__ att, float* __restrict__ scores) {
    // B chunk stored K-transposed: Bl[d*40 + k], pitch 40 halves -> 80B rows (16B aligned)
    __shared__ __align__(16) _Float16 Bl[D_EMB * 40];      // 20.0 KB
    __shared__ __align__(16) float    A_s[D_EMB];          // attention vector A[d]
    __shared__ float logits[L_OUT];                        // 8 KB
    __shared__ float red[256];
    __shared__ float s_w2[DA];

    const int tid   = threadIdx.x;
    const int b     = blockIdx.x;
    const int lane  = tid & 31;
    const int wave  = tid >> 5;
    const int mtile = wave & 3;
    const int ntb   = (wave >> 2) * 8;
    const int n     = lane & 15;              // column within 16x16 tile
    const int hi    = lane >> 4;              // half-wave selector (ISA K-split)
    const int* __restrict__ ib = idxb + (size_t)b * L_OUT;

    if (tid < D_EMB) A_s[tid] = 0.f;
    if (tid < DA)    s_w2[tid] = W2[tid];

    v8f acc[8] = {};

    const int gk    = tid & 31;               // gather: this thread's K row
    const int dbase = (tid >> 5) * 32;        // gather: this thread's D range

    for (int kc = 0; kc < NCHUNK; ++kc) {
        __syncthreads();                      // previous chunk's B reads done / A_s init visible
        {   // gather 32 embedding rows, f32 -> f16, transposed into LDS
            int l = kc * 32 + gk;
            if (l < L_OUT) {
                const float* row = emb + (size_t)ib[l] * D_EMB + dbase;
                #pragma unroll
                for (int j = 0; j < 32; ++j)
                    Bl[(dbase + j) * 40 + gk] = (_Float16)row[j];
            } else {
                #pragma unroll
                for (int j = 0; j < 32; ++j)
                    Bl[(dbase + j) * 40 + gk] = (_Float16)0.f;
            }
            int ln = kc * 32 + gk + 32;       // prefetch next chunk's row (global_prefetch_b8)
            if (ln < L_OUT)
                __builtin_prefetch(emb + (size_t)ib[ln] * D_EMB + dbase, 0, 0);
        }
        __syncthreads();

        // A fragment (16-bit 16x32 ISA layout): lane m = mtile*16+n,
        // elems 0..7 -> K = hi*8+e, elems 8..15 -> K = 16+hi*8+e
        const _Float16* w1r = W1h + (size_t)(mtile * 16 + n) * KPAD + kc * 32;
        h8 a0 = *(const h8*)(w1r + hi * 8);
        h8 a1 = *(const h8*)(w1r + 16 + hi * 8);
        v16h av;
        #pragma unroll
        for (int e = 0; e < 8; ++e) { av[e] = a0[e]; av[e + 8] = a1[e]; }

        // Preload ALL 8 B fragments (16x ds_load_b128) so the 8 WMMAs issue
        // back-to-back behind a single s_wait_dscnt instead of 8 stalls.
        v16h bvs[8];
        #pragma unroll
        for (int j = 0; j < 8; ++j) {
            // B fragment: lane column n, elems 0..7 -> K = hi*8+e, elems 8..15 -> K = 16+hi*8+e
            const _Float16* bp = &Bl[(size_t)((ntb + j) * 16 + n) * 40 + hi * 8];
            h8 b0 = *(const h8*)bp;
            h8 b1 = *(const h8*)(bp + 16);
            #pragma unroll
            for (int e = 0; e < 8; ++e) { bvs[j][e] = b0[e]; bvs[j][e + 8] = b1[e]; }
        }
        #pragma unroll
        for (int j = 0; j < 8; ++j)
            acc[j] = __builtin_amdgcn_wmma_f32_16x16x32_f16(
                false, av, false, bvs[j], (short)0, acc[j], false, false);
    }
    __syncthreads();

    // A[d] = sum_m W2[m] * tanh(t[m,d]); C/D layout: VGPR r -> M = mtile*16 + hi*8 + r
    #pragma unroll
    for (int j = 0; j < 8; ++j) {
        const int d = (ntb + j) * 16 + n;
        float p = 0.f;
        #pragma unroll
        for (int r = 0; r < 8; ++r)
            p += s_w2[mtile * 16 + hi * 8 + r] * tanhf(acc[j][r]);
        atomicAdd(&A_s[d], p);                // ds_add_f32
    }
    __syncthreads();

    // logits[l] = emb[idx[l]] . A   (rows are L2-resident: emb = 102MB < 192MB L2)
    float lmax = -1e30f;
    for (int l = tid; l < L_OUT; l += 256) {
        const float* row = emb + (size_t)ib[l] * D_EMB;
        float dot = 0.f;
        #pragma unroll
        for (int d = 0; d < D_EMB; d += 4) {
            float4 r4 = *(const float4*)(row + d);
            float4 a4 = *(const float4*)(&A_s[d]);
            dot += r4.x * a4.x + r4.y * a4.y + r4.z * a4.z + r4.w * a4.w;
        }
        logits[l] = dot;
        lmax = fmaxf(lmax, dot);
    }
    red[tid] = lmax; __syncthreads();
    for (int s = 128; s > 0; s >>= 1) { if (tid < s) red[tid] = fmaxf(red[tid], red[tid + s]); __syncthreads(); }
    const float gmax = red[0];
    __syncthreads();

    float lsum = 0.f;
    for (int l = tid; l < L_OUT; l += 256) {
        float e = expf(logits[l] - gmax);
        logits[l] = e;
        lsum += e;
    }
    red[tid] = lsum; __syncthreads();
    for (int s = 128; s > 0; s >>= 1) { if (tid < s) red[tid] += red[tid + s]; __syncthreads(); }
    const float inv = 1.0f / red[0];
    for (int l = tid; l < L_OUT; l += 256)
        att[(size_t)b * L_OUT + l] = logits[l] * inv;

    if (tid < SECTORS) {
        float s = lin_b[tid];
        #pragma unroll 8
        for (int d = 0; d < D_EMB; ++d) s += lin_w[tid * D_EMB + d] * A_s[d];
        scores[(size_t)b * SECTORS + tid] = s;
    }
}

// =============================== launch ===============================
extern "C" void kernel_launch(void* const* d_in, const int* in_sizes, int n_in,
                              void* d_out, int out_size, void* d_ws, size_t ws_size,
                              hipStream_t stream) {
    (void)in_sizes; (void)n_in; (void)out_size; (void)ws_size;
    const float* x     = (const float*)d_in[0];
    const float* w1    = (const float*)d_in[1];
    const float* b1    = (const float*)d_in[2];
    const float* gamma = (const float*)d_in[3];
    const float* beta  = (const float*)d_in[4];
    const float* w2    = (const float*)d_in[5];
    const float* b2    = (const float*)d_in[6];
    const float* emb   = (const float*)d_in[7];
    const float* W1    = (const float*)d_in[8];
    const float* W2    = (const float*)d_in[9];
    const float* lin_w = (const float*)d_in[10];
    const float* lin_b = (const float*)d_in[11];

    float*    ws    = (float*)d_ws;
    float*    h1    = ws;
    float*    stats = ws + STATS_FLOFF;
    int*      idxb  = (int*)((char*)d_ws + IDX_BYTE_OFF);
    _Float16* W1h   = (_Float16*)((char*)d_ws + W1H_BYTE_OFF);

    float* att    = (float*)d_out;
    float* scores = att + (size_t)BATCH * L_OUT;

    k_init<<<(DA * KPAD + 255) / 256, 256, 0, stream>>>(W1, W1h, stats);
    k_conv1_pool_stats<<<dim3(8, FILTERS, BATCH), 256, 0, stream>>>(x, w1, b1, h1, stats);
    k_bn_finalize<<<1, 64, 0, stream>>>(stats, gamma, beta, w2, b2);
    k_conv2_sigmoid_idx<<<dim3(8, BATCH), 256, 0, stream>>>(h1, w2, stats, idxb);
    k_attention<<<BATCH, 256, 0, stream>>>(emb, W2, lin_w, lin_b, idxb, W1h, att, scores);
}
